// Bert_CRF_68951404970498
// MI455X (gfx1250) — matrix operand order repaired and checked by reading, not compiled, not run on hardware
//
#include <hip/hip_runtime.h>
#include <hip/hip_bf16.h>
#include <math.h>

// Model constants (BERT-base + CRF from the reference)
constexpr int kNL  = 12;
constexpr int kH   = 768;
constexpr int kNH  = 12;
constexpr int kDH  = 64;
constexpr int kFF  = 3072;
constexpr int kB   = 8;
constexpr int kS   = 512;
constexpr int kLAB = 9;

typedef __bf16 bf16_t;
typedef __attribute__((ext_vector_type(16))) __bf16 v16bf;
typedef __attribute__((ext_vector_type(8)))  float  v8f;

union AFrag { v16bf v; bf16_t e[16]; uint4 q[2]; };
union CFrag { v8f   v; float  e[8];  };
union Pack4 { bf16_t h[4]; unsigned long long u64; };

// ---------------------------------------------------------------------------
// Weight packer: fp32 [K,N] row-major -> bf16 tile-fragment order.
// For each (kt,nt) 32x64 tile, emit the exact 4KB byte-image of the LDS
// fragment buffer Bst[8][32] (slabs 0..3 = elements 0..7, 4..7 = 8..15),
// so the GEMM can DMA it with global_load_async_to_lds_b128 untouched.
// Grid: (N/64, K/32, NL); per-layer stride K*N.
// ---------------------------------------------------------------------------
__global__ __launch_bounds__(128)
void pack_weights_k(const float* __restrict__ W, bf16_t* __restrict__ out,
                    int K, int N)
{
    int nt = blockIdx.x, kt = blockIdx.y, L = blockIdx.z;
    const float* src = W + (size_t)L * K * N;
    bf16_t* dst = out + (size_t)L * K * N +
                  ((size_t)nt * (K >> 5) + kt) * 2048;
    #pragma unroll
    for (int i = 0; i < 16; ++i) {
        int idx = i * 128 + threadIdx.x;   // 0..2047
        int kk = idx >> 6, n = idx & 63;
        float v = src[(size_t)(kt * 32 + kk) * N + nt * 64 + n];
        // fragment position: lane holds col n, kk = (lane>>4)*16 + e
        int slab = n >> 4;
        int lane = ((kk >> 4) << 4) + (n & 15);
        int e    = kk & 15;
        size_t off = (e < 8) ? ((size_t)(slab * 32 + lane) * 8 + e)
                             : (1024 + (size_t)(slab * 32 + lane) * 8 + (e - 8));
        dst[off] = (bf16_t)v;
    }
}

// ---------------------------------------------------------------------------
// Batched GEMM: C = act(alpha * A*B + bias), 64x64 block tile, K-step 32,
// 4 waves each owning a 32x32 quadrant = 2x2 v_wmma_f32_16x16x32_bf16.
//
// LDS tiles live in WMMA fragment order; fragment load = 2x ds_load_b128.
// PACKEDB: B is pre-packed bf16 -> staged with GLOBAL_LOAD_ASYNC_TO_LDS_B128
//          (ASYNCcnt), overlapping the DMA with the A-side fp32->bf16 work.
// Else:    B staged via fp32 loads + v_cvt_pk_bf16_f32 + ds stores.
//
// MODE 0: C[z*strideC + row*ldc + col]                (plain)
// MODE 1: QKV head split: row=(b,s), col=(head,d) -> [B,NH,S,DH]
// MODE 2: attention scores: + mask bias, C[z*S*S + row*S + col]
// MODE 3: context head merge: z=(b,head) -> [B,S,H]
// ---------------------------------------------------------------------------
template <int MODE, bool TRANSB, bool GELU, bool PACKEDB>
__global__ __launch_bounds__(128)
void gemm_bf16_wmma(const float* __restrict__ A, size_t strideA, int lda,
                    const float* __restrict__ Bm, size_t strideB, int ldb,
                    const float* __restrict__ bias,
                    float* __restrict__ C, size_t strideC, int ldc,
                    int M, int N, int K, float alpha,
                    const int* __restrict__ amask)
{
    // Fragment-order LDS: [slab 0..7][lane] 16-byte chunks (0..3=Lo, 4..7=Hi)
    __shared__ uint4 Ast[8][32];
    __shared__ uint4 Bst[8][32];

    const int tid  = threadIdx.x;
    const int lane = tid & 31;
    const int wave = tid >> 5;
    const int z    = blockIdx.z;
    const int bm   = blockIdx.y * 64;
    const int bn   = blockIdx.x * 64;

    const float* Ab = A  + (size_t)z * strideA;
    const float* Bp = Bm + (size_t)z * strideB;

    const int m0      = (wave >> 1) * 32;
    const int n0      = (wave & 1)  * 32;
    const int halfsel = lane >> 4;
    const int l15     = lane & 15;

    CFrag acc[2][2];
    #pragma unroll
    for (int i = 0; i < 2; ++i)
      #pragma unroll
      for (int j = 0; j < 2; ++j)
        #pragma unroll
        for (int r = 0; r < 8; ++r) acc[i][j].e[r] = 0.f;

    for (int kb = 0; kb < K; kb += 32) {
        // ---- Stage B tile first (async DMA path can overlap A work) ----
        if (PACKEDB) {
            const bf16_t* srcT = (const bf16_t*)Bm +
                ((size_t)blockIdx.x * (K >> 5) + (kb >> 5)) * 2048;
            unsigned lds  = (unsigned)(size_t)(&Bst[0][0]) + (unsigned)tid * 16u;
            unsigned long long ga =
                (unsigned long long)(size_t)srcT + (unsigned long long)tid * 16ull;
            // INST_OFFSET applies to both LDS and global address (ISA 15.18)
            asm volatile("global_load_async_to_lds_b128 %0, %1, off"
                         :: "v"(lds), "v"(ga) : "memory");
            asm volatile("global_load_async_to_lds_b128 %0, %1, off offset:2048"
                         :: "v"(lds), "v"(ga) : "memory");
        } else if (TRANSB) {
            // Source row-major [n][k]: vectorize along k, b64 stores.
            #pragma unroll
            for (int i = 0; i < 4; ++i) {
                int idx = i * 128 + tid;
                int n   = idx >> 3;          // 0..63
                int k4  = (idx & 7) << 2;    // 0..28
                float4 fv = *(const float4*)(Bp + (size_t)(bn + n) * ldb + kb + k4);
                Pack4 p;
                p.h[0] = (bf16_t)fv.x; p.h[1] = (bf16_t)fv.y;
                p.h[2] = (bf16_t)fv.z; p.h[3] = (bf16_t)fv.w;
                int slab = n >> 4;
                int ld   = ((k4 >> 4) << 4) + (n & 15);
                int e0   = k4 & 15;
                uint4* chunk = (e0 < 8) ? &Bst[slab][ld] : &Bst[4 + slab][ld];
                *(unsigned long long*)((bf16_t*)chunk + (e0 & 7)) = p.u64;
            }
        } else {
            // Source row-major [k][n]: vectorize along n, 4-way b16 scatter.
            #pragma unroll
            for (int i = 0; i < 4; ++i) {
                int idx = i * 128 + tid;
                int kk  = idx >> 4;          // 0..31
                int n4  = (idx & 15) << 2;   // 0..60
                float4 fv = *(const float4*)(Bp + (size_t)(kb + kk) * ldb + bn + n4);
                bf16_t h[4] = {(bf16_t)fv.x, (bf16_t)fv.y, (bf16_t)fv.z, (bf16_t)fv.w};
                int e        = kk & 15;
                int laneBase = (kk >> 4) << 4;
                #pragma unroll
                for (int d = 0; d < 4; ++d) {
                    int n = n4 + d;
                    uint4* chunk = (e < 8) ? &Bst[n >> 4][laneBase + (n & 15)]
                                           : &Bst[4 + (n >> 4)][laneBase + (n & 15)];
                    ((bf16_t*)chunk)[e & 7] = h[d];
                }
            }
        }

        // ---- Stage A tile (64 rows x 32 k): 512 float4, 4 per thread ----
        #pragma unroll
        for (int i = 0; i < 4; ++i) {
            int idx = i * 128 + tid;
            int r   = idx >> 3;          // 0..63
            int c4  = (idx & 7) << 2;    // 0,4,...,28
            float4 fv = *(const float4*)(Ab + (size_t)(bm + r) * lda + kb + c4);
            Pack4 p;
            p.h[0] = (bf16_t)fv.x; p.h[1] = (bf16_t)fv.y;
            p.h[2] = (bf16_t)fv.z; p.h[3] = (bf16_t)fv.w;
            // contiguous A-layout: kk -> (half = kk>>4, e = kk&15)
            int slab = r >> 4, m = r & 15;
            int half = c4 >> 4;
            int e0   = c4 & 15;
            int ld   = half * 16 + m;
            uint4* chunk = (e0 < 8) ? &Ast[slab][ld] : &Ast[4 + slab][ld];
            *(unsigned long long*)((bf16_t*)chunk + (e0 & 7)) = p.u64;
        }

        if (PACKEDB)
            asm volatile("s_wait_asynccnt 0x0" ::: "memory");
        __syncthreads();

        // ---- Fragment loads: 2x ds_load_b128 per fragment ----
        AFrag af[2], bfr[2];
        const int as0 = m0 >> 4, bs0 = n0 >> 4;
        af[0].q[0]  = Ast[as0][lane];         af[0].q[1]  = Ast[4 + as0][lane];
        af[1].q[0]  = Ast[as0 + 1][lane];     af[1].q[1]  = Ast[5 + as0][lane];
        bfr[0].q[0] = Bst[bs0][lane];         bfr[0].q[1] = Bst[4 + bs0][lane];
        bfr[1].q[0] = Bst[bs0 + 1][lane];     bfr[1].q[1] = Bst[5 + bs0][lane];

        #pragma unroll
        for (int i = 0; i < 2; ++i)
          #pragma unroll
          for (int j = 0; j < 2; ++j)
            acc[i][j].v = __builtin_amdgcn_wmma_f32_16x16x32_bf16(
                false, af[i].v, false, bfr[j].v, (short)0, acc[i][j].v,
                false, false);
        __syncthreads();
    }

    // ---- Epilogue: C/D lane layout: N = lane&15, rows (lane>>4)*8 + r ----
    #pragma unroll
    for (int i = 0; i < 2; ++i) {
      #pragma unroll
      for (int j = 0; j < 2; ++j) {
        #pragma unroll
        for (int r = 0; r < 8; ++r) {
            int row = bm + m0 + i * 16 + halfsel * 8 + r;
            int col = bn + n0 + j * 16 + l15;
            if (row >= M || col >= N) continue;
            float v = acc[i][j].e[r] * alpha;
            if (bias) v += bias[col];
            if (GELU) v = 0.5f * v * (1.f + erff(v * 0.70710678f));
            size_t off;
            if (MODE == 0) {
                off = (size_t)z * strideC + (size_t)row * ldc + col;
            } else if (MODE == 1) {
                int b = row / kS, s = row % kS, hd = col / kDH, d = col % kDH;
                off = (((size_t)(b * kNH + hd)) * kS + s) * kDH + d;
            } else if (MODE == 2) {
                int b = z / kNH;
                v += (1.f - (float)amask[b * kS + col]) * -10000.f;
                off = (size_t)z * strideC + (size_t)row * kS + col;
            } else { // MODE 3
                int b = z / kNH, hd = z % kNH;
                off = ((size_t)b * kS + row) * kH + hd * kDH + col;
            }
            C[off] = v;
        }
      }
    }
}

// ---------------------------------------------------------------------------
// Embedding gather + LayerNorm. One block per token (768 = 3*256).
// ---------------------------------------------------------------------------
__global__ __launch_bounds__(256)
void embed_ln_k(const float* __restrict__ we, const float* __restrict__ pe,
                const float* __restrict__ te,
                const float* __restrict__ g, const float* __restrict__ bta,
                const int* __restrict__ ids, const int* __restrict__ tt,
                float* __restrict__ out)
{
    __shared__ float red[256];
    int tok = blockIdx.x;
    int s   = tok % kS;
    int id  = ids[tok];
    int ty  = tt[tok];
    float vals[3];
    float sum = 0.f;
    #pragma unroll
    for (int i = 0; i < 3; ++i) {
        int j = threadIdx.x + i * 256;
        float v = we[(size_t)id * kH + j] + pe[(size_t)s * kH + j] +
                  te[(size_t)ty * kH + j];
        vals[i] = v; sum += v;
    }
    red[threadIdx.x] = sum; __syncthreads();
    for (int st = 128; st > 0; st >>= 1) {
        if (threadIdx.x < st) red[threadIdx.x] += red[threadIdx.x + st];
        __syncthreads();
    }
    float mu = red[0] * (1.f / kH); __syncthreads();
    float vs = 0.f;
    #pragma unroll
    for (int i = 0; i < 3; ++i) { float d = vals[i] - mu; vs += d * d; }
    red[threadIdx.x] = vs; __syncthreads();
    for (int st = 128; st > 0; st >>= 1) {
        if (threadIdx.x < st) red[threadIdx.x] += red[threadIdx.x + st];
        __syncthreads();
    }
    float rstd = rsqrtf(red[0] * (1.f / kH) + 1e-12f);
    #pragma unroll
    for (int i = 0; i < 3; ++i) {
        int j = threadIdx.x + i * 256;
        out[(size_t)tok * kH + j] = (vals[i] - mu) * rstd * g[j] + bta[j];
    }
}

// ---------------------------------------------------------------------------
// Residual add + LayerNorm (in-place on h). One block per token.
// ---------------------------------------------------------------------------
__global__ __launch_bounds__(256)
void add_ln_k(float* __restrict__ h, const float* __restrict__ dlt,
              const float* __restrict__ g, const float* __restrict__ bta)
{
    __shared__ float red[256];
    int tok = blockIdx.x;
    float vals[3];
    float sum = 0.f;
    #pragma unroll
    for (int i = 0; i < 3; ++i) {
        int j = threadIdx.x + i * 256;
        float v = h[(size_t)tok * kH + j] + dlt[(size_t)tok * kH + j];
        vals[i] = v; sum += v;
    }
    red[threadIdx.x] = sum; __syncthreads();
    for (int st = 128; st > 0; st >>= 1) {
        if (threadIdx.x < st) red[threadIdx.x] += red[threadIdx.x + st];
        __syncthreads();
    }
    float mu = red[0] * (1.f / kH); __syncthreads();
    float vs = 0.f;
    #pragma unroll
    for (int i = 0; i < 3; ++i) { float d = vals[i] - mu; vs += d * d; }
    red[threadIdx.x] = vs; __syncthreads();
    for (int st = 128; st > 0; st >>= 1) {
        if (threadIdx.x < st) red[threadIdx.x] += red[threadIdx.x + st];
        __syncthreads();
    }
    float rstd = rsqrtf(red[0] * (1.f / kH) + 1e-12f);
    #pragma unroll
    for (int i = 0; i < 3; ++i) {
        int j = threadIdx.x + i * 256;
        h[(size_t)tok * kH + j] = (vals[i] - mu) * rstd * g[j] + bta[j];
    }
}

// ---------------------------------------------------------------------------
// Row softmax over S=512 (mask bias already folded in). One block per row.
// ---------------------------------------------------------------------------
__global__ __launch_bounds__(256)
void softmax_rows_k(float* __restrict__ p)
{
    __shared__ float red[256];
    size_t base = (size_t)blockIdx.x * kS;
    float a = p[base + threadIdx.x];
    float b = p[base + threadIdx.x + 256];
    red[threadIdx.x] = fmaxf(a, b); __syncthreads();
    for (int st = 128; st > 0; st >>= 1) {
        if (threadIdx.x < st)
            red[threadIdx.x] = fmaxf(red[threadIdx.x], red[threadIdx.x + st]);
        __syncthreads();
    }
    float m = red[0]; __syncthreads();
    float ea = __expf(a - m), eb = __expf(b - m);
    red[threadIdx.x] = ea + eb; __syncthreads();
    for (int st = 128; st > 0; st >>= 1) {
        if (threadIdx.x < st) red[threadIdx.x] += red[threadIdx.x + st];
        __syncthreads();
    }
    float inv = 1.f / red[0];
    p[base + threadIdx.x]       = ea * inv;
    p[base + threadIdx.x + 256] = eb * inv;
}

// ---------------------------------------------------------------------------
// Classifier: feats = seq @ clf_W + clf_b (N=9, too small for a WMMA tile).
// ---------------------------------------------------------------------------
__global__ __launch_bounds__(32)
void classifier_k(const float* __restrict__ h, const float* __restrict__ W,
                  const float* __restrict__ b, float* __restrict__ feats)
{
    int row = blockIdx.x;
    int l = threadIdx.x;
    if (l < kLAB) {
        float s = b[l];
        const float* hr = h + (size_t)row * kH;
        for (int k = 0; k < kH; ++k) s += hr[k] * W[k * kLAB + l];
        feats[(size_t)row * kLAB + l] = s;
    }
}

// ---------------------------------------------------------------------------
// CRF negative log-likelihood (mean over batch). Single block, sequential S.
// ---------------------------------------------------------------------------
__global__ __launch_bounds__(128)
void crf_forward_k(const float* __restrict__ feats, const int* __restrict__ tags,
                   const int* __restrict__ mask,
                   const float* __restrict__ stt, const float* __restrict__ endd,
                   const float* __restrict__ tr, float* __restrict__ out_loss)
{
    __shared__ float sc[2][kB][kLAB];
    __shared__ float logZ[kB], num[kB];
    int tid = threadIdx.x;
    int b = tid / kLAB, j = tid % kLAB;
    if (tid < kB * kLAB)
        sc[0][b][j] = stt[j] + feats[((size_t)b * kS) * kLAB + j];
    __syncthreads();
    int cur = 0;
    for (int t = 1; t < kS; ++t) {
        if (tid < kB * kLAB) {
            float mx = -1e30f;
            for (int i = 0; i < kLAB; ++i)
                mx = fmaxf(mx, sc[cur][b][i] + tr[i * kLAB + j]);
            float s = 0.f;
            for (int i = 0; i < kLAB; ++i)
                s += __expf(sc[cur][b][i] + tr[i * kLAB + j] - mx);
            float nxt = mx + __logf(s) + feats[((size_t)b * kS + t) * kLAB + j];
            sc[cur ^ 1][b][j] = (mask[b * kS + t] > 0) ? nxt : sc[cur][b][j];
        }
        __syncthreads();
        cur ^= 1;
    }
    if (tid < kB) {
        int bb = tid;
        float mx = -1e30f;
        for (int jj = 0; jj < kLAB; ++jj)
            mx = fmaxf(mx, sc[cur][bb][jj] + endd[jj]);
        float s = 0.f;
        for (int jj = 0; jj < kLAB; ++jj)
            s += __expf(sc[cur][bb][jj] + endd[jj] - mx);
        logZ[bb] = mx + __logf(s);
        int t0 = tags[bb * kS];
        float n = stt[t0] + feats[((size_t)bb * kS) * kLAB + t0];
        int prev = t0;
        for (int t = 1; t < kS; ++t) {
            int tg = tags[bb * kS + t];
            if (mask[bb * kS + t] > 0)
                n += feats[((size_t)bb * kS + t) * kLAB + tg] + tr[prev * kLAB + tg];
            prev = tg;
        }
        int sm = 0;
        for (int t = 0; t < kS; ++t) sm += (mask[bb * kS + t] > 0) ? 1 : 0;
        n += endd[tags[bb * kS + (sm - 1)]];
        num[bb] = n;
    }
    __syncthreads();
    if (tid == 0) {
        float l = 0.f;
        for (int bb = 0; bb < kB; ++bb) l += logZ[bb] - num[bb];
        out_loss[0] = l / kB;
    }
}

// ---------------------------------------------------------------------------
// Viterbi decode. Single block, sequential forward + per-batch backtrack.
// ---------------------------------------------------------------------------
__global__ __launch_bounds__(128)
void viterbi_k(const float* __restrict__ feats, const int* __restrict__ mask,
               const float* __restrict__ stt, const float* __restrict__ endd,
               const float* __restrict__ tr,
               int* __restrict__ hist, float* __restrict__ out_tags)
{
    __shared__ float sc[2][kB][kLAB];
    int tid = threadIdx.x;
    int b = tid / kLAB, j = tid % kLAB;
    if (tid < kB * kLAB)
        sc[0][b][j] = stt[j] + feats[((size_t)b * kS) * kLAB + j];
    __syncthreads();
    int cur = 0;
    for (int t = 1; t < kS; ++t) {
        if (tid < kB * kLAB) {
            float best = -1e30f; int bi = 0;
            for (int i = 0; i < kLAB; ++i) {
                float c = sc[cur][b][i] + tr[i * kLAB + j];
                if (c > best) { best = c; bi = i; }
            }
            hist[(((t - 1) * kB) + b) * kLAB + j] = bi;
            float nxt = best + feats[((size_t)b * kS + t) * kLAB + j];
            sc[cur ^ 1][b][j] = (mask[b * kS + t] > 0) ? nxt : sc[cur][b][j];
        }
        __syncthreads();
        cur ^= 1;
    }
    if (tid < kB) {
        int bb = tid;
        float best = -1e30f; int bj = 0;
        for (int jj = 0; jj < kLAB; ++jj) {
            float c = sc[cur][bb][jj] + endd[jj];
            if (c > best) { best = c; bj = jj; }
        }
        int curt = bj;
        out_tags[bb * kS + (kS - 1)] = (float)curt;
        for (int t = kS - 2; t >= 0; --t) {
            curt = hist[((t * kB) + bb) * kLAB + curt];
            out_tags[bb * kS + t] = (float)curt;
        }
    }
}

// ---------------------------------------------------------------------------
extern "C" void kernel_launch(void* const* d_in, const int* in_sizes, int n_in,
                              void* d_out, int out_size, void* d_ws, size_t ws_size,
                              hipStream_t stream)
{
    (void)in_sizes; (void)n_in; (void)out_size; (void)ws_size;

    const float* word_emb  = (const float*)d_in[0];
    const float* pos_emb   = (const float*)d_in[1];
    const float* type_emb  = (const float*)d_in[2];
    const float* emb_ln_s  = (const float*)d_in[3];
    const float* emb_ln_b  = (const float*)d_in[4];
    const float* Wq        = (const float*)d_in[5];
    const float* bq        = (const float*)d_in[6];
    const float* Wk        = (const float*)d_in[7];
    const float* bk        = (const float*)d_in[8];
    const float* Wv        = (const float*)d_in[9];
    const float* bv        = (const float*)d_in[10];
    const float* Wo        = (const float*)d_in[11];
    const float* bo        = (const float*)d_in[12];
    const float* ln1_s     = (const float*)d_in[13];
    const float* ln1_b     = (const float*)d_in[14];
    const float* W1        = (const float*)d_in[15];
    const float* b1        = (const float*)d_in[16];
    const float* W2        = (const float*)d_in[17];
    const float* b2        = (const float*)d_in[18];
    const float* ln2_s     = (const float*)d_in[19];
    const float* ln2_b     = (const float*)d_in[20];
    const float* clf_W     = (const float*)d_in[21];
    const float* clf_b     = (const float*)d_in[22];
    const float* crf_start = (const float*)d_in[23];
    const float* crf_end   = (const float*)d_in[24];
    const float* crf_trans = (const float*)d_in[25];
    const int*   input_ids = (const int*)d_in[26];
    const int*   type_ids  = (const int*)d_in[27];
    const int*   amask     = (const int*)d_in[28];
    const int*   labels    = (const int*)d_in[29];
    float*       out       = (float*)d_out;

    // Workspace bump allocator (floats, then ints, then bf16 pack region)
    float* ws = (float*)d_ws;
    const size_t BSH = (size_t)kB * kS * kH;
    const size_t BSF = (size_t)kB * kS * kFF;
    const size_t SCN = (size_t)kB * kNH * kS * kS;
    float* hbuf   = ws; ws += BSH;
    float* tmp    = ws; ws += BSH;
    float* qbuf   = ws; ws += BSH;   // q heads; later reused for context
    float* kbuf   = ws; ws += BSH;
    float* vbuf   = ws; ws += BSH;
    float* ffb    = ws; ws += BSF;
    float* scores = ws; ws += SCN;
    float* feats  = ws; ws += (size_t)kB * kS * kLAB;
    int*   hist   = (int*)ws; ws += (size_t)(kS - 1) * kB * kLAB;  // int area

    bf16_t* wp = (bf16_t*)ws;
    const size_t WHH = (size_t)kNL * kH * kH;
    const size_t WHF = (size_t)kNL * kH * kFF;
    bf16_t* wpQ = wp; wp += WHH;
    bf16_t* wpK = wp; wp += WHH;
    bf16_t* wpV = wp; wp += WHH;
    bf16_t* wpO = wp; wp += WHH;
    bf16_t* wp1 = wp; wp += WHF;
    bf16_t* wp2 = wp; wp += WHF;

    const dim3 blk(128);
    const int  Mtok = kB * kS;  // 4096 token rows

    // 0) Pack all weights fp32 -> bf16 tile-fragment order (DMA-ready)
    {
        dim3 gp(kH / 64, kH / 32, kNL);
        pack_weights_k<<<gp, blk, 0, stream>>>(Wq, wpQ, kH, kH);
        pack_weights_k<<<gp, blk, 0, stream>>>(Wk, wpK, kH, kH);
        pack_weights_k<<<gp, blk, 0, stream>>>(Wv, wpV, kH, kH);
        pack_weights_k<<<gp, blk, 0, stream>>>(Wo, wpO, kH, kH);
        dim3 gp1(kFF / 64, kH / 32, kNL);
        pack_weights_k<<<gp1, blk, 0, stream>>>(W1, wp1, kH, kFF);
        dim3 gp2(kH / 64, kFF / 32, kNL);
        pack_weights_k<<<gp2, blk, 0, stream>>>(W2, wp2, kFF, kH);
    }

    // 1) Embedding + LN
    embed_ln_k<<<Mtok, 256, 0, stream>>>(word_emb, pos_emb, type_emb,
                                         emb_ln_s, emb_ln_b,
                                         input_ids, type_ids, hbuf);

    // 2) Encoder layers
    for (int L = 0; L < kNL; ++L) {
        const float* pQ = (const float*)(wpQ + (size_t)L * kH * kH);
        const float* pK = (const float*)(wpK + (size_t)L * kH * kH);
        const float* pV = (const float*)(wpV + (size_t)L * kH * kH);
        const float* pO = (const float*)(wpO + (size_t)L * kH * kH);
        const float* p1 = (const float*)(wp1 + (size_t)L * kH * kFF);
        const float* p2 = (const float*)(wp2 + (size_t)L * kFF * kH);
        const float* bq_ = bq + (size_t)L * kH;
        const float* bk_ = bk + (size_t)L * kH;
        const float* bv_ = bv + (size_t)L * kH;
        const float* bo_ = bo + (size_t)L * kH;
        const float* b1_ = b1 + (size_t)L * kFF;
        const float* b2_ = b2 + (size_t)L * kH;

        // QKV projections -> [B, NH, S, DH]  (async-DMA packed weights)
        dim3 gproj(kH / 64, Mtok / 64, 1);
        gemm_bf16_wmma<1, false, false, true><<<gproj, blk, 0, stream>>>(
            hbuf, 0, kH, pQ, 0, kH, bq_, qbuf, 0, kH, Mtok, kH, kH, 1.f, nullptr);
        gemm_bf16_wmma<1, false, false, true><<<gproj, blk, 0, stream>>>(
            hbuf, 0, kH, pK, 0, kH, bk_, kbuf, 0, kH, Mtok, kH, kH, 1.f, nullptr);
        gemm_bf16_wmma<1, false, false, true><<<gproj, blk, 0, stream>>>(
            hbuf, 0, kH, pV, 0, kH, bv_, vbuf, 0, kH, Mtok, kH, kH, 1.f, nullptr);

        // scores = q @ k^T / sqrt(DH) + mask_bias (batched over B*NH)
        dim3 gsc(kS / 64, kS / 64, kB * kNH);
        gemm_bf16_wmma<2, true, false, false><<<gsc, blk, 0, stream>>>(
            qbuf, (size_t)kS * kDH, kDH,
            kbuf, (size_t)kS * kDH, kDH, nullptr,
            scores, (size_t)kS * kS, kS,
            kS, kS, kDH, 0.125f, amask);

        softmax_rows_k<<<kB * kNH * kS, 256, 0, stream>>>(scores);

        // ctx = P @ v, merge heads -> [B, S, H] (reuses qbuf)
        dim3 gctx(kDH / 64, kS / 64, kB * kNH);
        gemm_bf16_wmma<3, false, false, false><<<gctx, blk, 0, stream>>>(
            scores, (size_t)kS * kS, kS,
            vbuf, (size_t)kS * kDH, kDH, nullptr,
            qbuf, 0, 0,
            kS, kDH, kS, 1.f, nullptr);

        // output projection
        gemm_bf16_wmma<0, false, false, true><<<gproj, blk, 0, stream>>>(
            qbuf, 0, kH, pO, 0, kH, bo_, tmp, 0, kH, Mtok, kH, kH, 1.f, nullptr);
        add_ln_k<<<Mtok, 256, 0, stream>>>(hbuf, tmp, ln1_s + (size_t)L * kH,
                                           ln1_b + (size_t)L * kH);

        // FFN
        dim3 gff1(kFF / 64, Mtok / 64, 1);
        gemm_bf16_wmma<0, false, true, true><<<gff1, blk, 0, stream>>>(
            hbuf, 0, kH, p1, 0, kFF, b1_, ffb, 0, kFF, Mtok, kFF, kH, 1.f, nullptr);
        dim3 gff2(kH / 64, Mtok / 64, 1);
        gemm_bf16_wmma<0, false, false, true><<<gff2, blk, 0, stream>>>(
            ffb, 0, kFF, p2, 0, kH, b2_, tmp, 0, kH, Mtok, kH, kFF, 1.f, nullptr);
        add_ln_k<<<Mtok, 256, 0, stream>>>(hbuf, tmp, ln2_s + (size_t)L * kH,
                                           ln2_b + (size_t)L * kH);
    }

    // 3) Classifier
    classifier_k<<<Mtok, 32, 0, stream>>>(hbuf, clf_W, clf_b, feats);

    // 4) CRF loss -> out[0]
    crf_forward_k<<<1, 128, 0, stream>>>(feats, labels, amask,
                                         crf_start, crf_end, crf_trans, out);

    // 5) Viterbi tags -> out[1 .. 1+B*S)
    viterbi_k<<<1, 128, 0, stream>>>(feats, amask, crf_start, crf_end, crf_trans,
                                     hist, out + 1);
}
